// Attention_55009941128016
// MI455X (gfx1250) — matrix operand order repaired and checked by reading, not compiled, not run on hardware
//
#include <hip/hip_runtime.h>
#include <hip/hip_bf16.h>
#include <math.h>

// ---------------- problem constants ----------------
#define HEADS  16
#define DHEAD  64
#define DIM    1024
#define INNER  1024
#define NQKV   3072
#define BATCH  4
#define SEQ    2048
#define ROWS   (BATCH * SEQ)   // 8192
#define EPS    1e-5f
#define KTILE  64              // flash-attn key tile staged in LDS

// ---------------- WMMA fragment types (wave32, gfx1250) ----------------
typedef __attribute__((ext_vector_type(16))) __bf16        v16bf;
typedef __attribute__((ext_vector_type(8)))  float          v8f;
typedef __attribute__((ext_vector_type(8)))  unsigned short v8us;

union FragBF {
    v16bf bf;
    v8us  h[2];
    unsigned short u[16];
};

__device__ __forceinline__ unsigned short f2bf(float f) {
    unsigned u = __float_as_uint(f);
    unsigned r = u + 0x7FFFu + ((u >> 16) & 1u);   // round-to-nearest-even
    return (unsigned short)(r >> 16);
}
__device__ __forceinline__ float bf2f(unsigned short s) {
    return __uint_as_float(((unsigned)s) << 16);
}

// ---------------- Tensor Data Mover (TDM) helpers ----------------
#if defined(__has_builtin)
#if __has_builtin(__builtin_amdgcn_tensor_load_to_lds)
#define HAVE_TDM 1
#endif
#endif

#ifdef HAVE_TDM
#pragma message("CDNA5: tensor_load_to_lds (TDM) path enabled")
typedef __attribute__((ext_vector_type(4))) unsigned int u32x4;
typedef __attribute__((ext_vector_type(4))) int          i32x4;
typedef __attribute__((ext_vector_type(8))) int          i32x8;

__device__ __forceinline__ unsigned lds_addr_of(const void* p) {
    // LDS aperture: addr[31:0] is the LDS byte offset (ISA 10.2)
    return (unsigned)(unsigned long long)p;
}

// 2D tile load: tileW elements (2B each) x tileH rows, row stride strideElems.
// Builds D# groups per CDNA5 ISA (§8.3/8.4); groups 2/3 zero (2D).
__device__ __forceinline__ void tdm_load_2d(const void* gptr, unsigned ldsAddr,
                                            unsigned tileW, unsigned tileH,
                                            unsigned strideElems) {
    unsigned long long ga = (unsigned long long)gptr;
    u32x4 g0;
    g0[0] = 1u;                                         // count=1, user mode
    g0[1] = ldsAddr;                                    // lds_addr [63:32]
    g0[2] = (unsigned)ga;                               // global_addr [95:64]
    g0[3] = (unsigned)((ga >> 32) & 0x1FFFFFFull) | (2u << 30); // addr hi, type=2
    i32x8 g1;
    g1[0] = 0x10000;                                    // data_size=1 (2 bytes)
    g1[1] = (int)(tileW << 16);                         // tensor_dim0 lo -> bits 63:48
    g1[2] = (int)(tileH << 16);                         // dim0 hi=0 | tensor_dim1 lo
    g1[3] = (int)(tileW << 16);                         // dim1 hi=0 | tile_dim0
    g1[4] = (int)(tileH & 0xFFFF);                      // tile_dim1 | tile_dim2=0
    g1[5] = (int)strideElems;                           // tensor_dim0_stride lo32
    g1[6] = 0;
    g1[7] = 0;
    i32x4 z = {0, 0, 0, 0};
#if __clang_major__ >= 23
    i32x8 z8 = {};
    __builtin_amdgcn_tensor_load_to_lds(g0, g1, z, z, z8, 0);
#else
    __builtin_amdgcn_tensor_load_to_lds(g0, g1, z, z, 0);
#endif
}

__device__ __forceinline__ void wait_tensorcnt_0() {
#if __has_builtin(__builtin_amdgcn_s_wait_tensorcnt)
    __builtin_amdgcn_s_wait_tensorcnt(0);
#else
    asm volatile("s_wait_tensorcnt 0" ::: "memory");
#endif
}
__device__ __forceinline__ void wait_tensorcnt_2() {
#if __has_builtin(__builtin_amdgcn_s_wait_tensorcnt)
    __builtin_amdgcn_s_wait_tensorcnt(2);
#else
    asm volatile("s_wait_tensorcnt 2" ::: "memory");
#endif
}
#else
#pragma message("CDNA5: TDM builtin unavailable -> cooperative LDS staging fallback")
#endif  // HAVE_TDM

// ---------------- 1. LayerNorm -> bf16 ----------------
__global__ __launch_bounds__(256)
void ln_kernel(const float* __restrict__ x, const float* __restrict__ g,
               const float* __restrict__ b, unsigned short* __restrict__ xn) {
    int row = blockIdx.x;
    const float* xr = x + (size_t)row * DIM;
    float v[4];
    float s = 0.f;
#pragma unroll
    for (int i = 0; i < 4; ++i) { v[i] = xr[threadIdx.x + i * 256]; s += v[i]; }
    __shared__ float red[256];
    red[threadIdx.x] = s; __syncthreads();
    for (int off = 128; off > 0; off >>= 1) {
        if (threadIdx.x < (unsigned)off) red[threadIdx.x] += red[threadIdx.x + off];
        __syncthreads();
    }
    float mu = red[0] * (1.f / DIM);
    __syncthreads();
    float s2 = 0.f;
#pragma unroll
    for (int i = 0; i < 4; ++i) { float d = v[i] - mu; s2 += d * d; }
    red[threadIdx.x] = s2; __syncthreads();
    for (int off = 128; off > 0; off >>= 1) {
        if (threadIdx.x < (unsigned)off) red[threadIdx.x] += red[threadIdx.x + off];
        __syncthreads();
    }
    float rs = rsqrtf(red[0] * (1.f / DIM) + EPS);
#pragma unroll
    for (int i = 0; i < 4; ++i) {
        int c = threadIdx.x + i * 256;
        xn[(size_t)row * DIM + c] = f2bf((v[i] - mu) * rs * g[c] + b[c]);
    }
}

// ---------------- 2. weight transpose+convert, RoPE tables ----------------
__global__ __launch_bounds__(256)
void conv_wqkv(const float* __restrict__ w, unsigned short* __restrict__ wt) {
    int id = blockIdx.x * 256 + threadIdx.x;    // over 3072*1024, wt[n][k]
    int n = id >> 10, k = id & 1023;
    wt[id] = f2bf(w[(size_t)k * NQKV + n]);
}
__global__ __launch_bounds__(256)
void conv_wout(const float* __restrict__ w, unsigned short* __restrict__ wt) {
    int id = blockIdx.x * 256 + threadIdx.x;    // over 1024*1024, wt[n][k]
    int n = id >> 10, k = id & 1023;
    wt[id] = f2bf(w[(size_t)k * DIM + n]);
}
__global__ __launch_bounds__(256)
void rope_tables_kernel(float* __restrict__ cosT, float* __restrict__ sinT) {
    int id = blockIdx.x * 256 + threadIdx.x;    // over 2048*32
    int n = id >> 5, j = id & 31;
    float inv = __expf(-0.28782313662425572f * (float)j);  // 10000^(-j/32)
    float a = (float)n * inv;
    cosT[id] = cosf(a);
    sinT[id] = sinf(a);
}

// ---------------- 3. QKV GEMM (WMMA bf16, 64Mx64N/wave) + RoPE epilogue ----------------
// A: xn [8192][1024] bf16, B: wqkvT [3072][1024] bf16 (N-major)
__global__ __launch_bounds__(128)
void qkv_gemm(const unsigned short* __restrict__ A,
              const unsigned short* __restrict__ Bt,
              const float* __restrict__ cosT, const float* __restrict__ sinT,
              unsigned short* __restrict__ qO,   // [b,h,n,d]
              unsigned short* __restrict__ kO,   // [b,h,n,d]
              unsigned short* __restrict__ vtO)  // [b,h,d,n]
{
    int wave  = threadIdx.x >> 5;
    int lane  = threadIdx.x & 31;
    int laneN = lane & 15;
    int hi    = lane >> 4;
    int m0    = blockIdx.y * 64;
    int nbase = blockIdx.x * 256 + wave * 64;

    v8f acc[4][4] = {};                         // [mTile][nTile]
    const unsigned short* arow[4];
#pragma unroll
    for (int mt = 0; mt < 4; ++mt)
        arow[mt] = A + (size_t)(m0 + mt * 16 + laneN) * DIM;
    const unsigned short* brow[4];
#pragma unroll
    for (int a = 0; a < 4; ++a)
        brow[a] = Bt + (size_t)(nbase + a * 16 + laneN) * DIM + hi * 8;
    int aofs = hi * 8;

    for (int k = 0; k < DIM; k += 32) {
        FragBF fa[4];
#pragma unroll
        for (int mt = 0; mt < 4; ++mt) {
            fa[mt].h[0] = *(const v8us*)(arow[mt] + k + aofs);
            fa[mt].h[1] = *(const v8us*)(arow[mt] + k + 16 + aofs);
        }
#pragma unroll
        for (int a = 0; a < 4; ++a) {
            FragBF fb;
            fb.h[0] = *(const v8us*)(brow[a] + k);
            fb.h[1] = *(const v8us*)(brow[a] + k + 16);
#pragma unroll
            for (int mt = 0; mt < 4; ++mt)
                acc[mt][a] = __builtin_amdgcn_wmma_f32_16x16x32_bf16(
                    false, fa[mt].bf, false, fb.bf, (short)0, acc[mt][a], false, false);
        }
    }

    int region = nbase >> 10;         // 0=q, 1=k, 2=v (wave never straddles)
    int chead  = nbase & 1023;
    int h      = chead >> 6;          // wave spans exactly one head

    if (region < 2) {
        unsigned short* dst = (region == 0) ? qO : kO;
#pragma unroll
        for (int mt = 0; mt < 4; ++mt)
#pragma unroll
            for (int r = 0; r < 8; ++r) {
                int gRow = m0 + mt * 16 + hi * 8 + r;
                int bb = gRow >> 11, n = gRow & 2047;
#pragma unroll
                for (int a = 0; a < 4; ++a) {
                    int d = a * 16 + laneN;
                    int j = d >> 1;
                    float c = cosT[n * 32 + j], s = sinT[n * 32 + j];
                    float xv  = acc[mt][a][r];
                    float rot = (a < 2) ? -acc[mt][a + 2][r] : acc[mt][a - 2][r];
                    dst[((size_t)(bb * HEADS + h) * SEQ + n) * DHEAD + d] =
                        f2bf(xv * c + rot * s);
                }
            }
    } else {
#pragma unroll
        for (int mt = 0; mt < 4; ++mt)
#pragma unroll
            for (int r = 0; r < 8; ++r) {
                int gRow = m0 + mt * 16 + hi * 8 + r;
                int bb = gRow >> 11, n = gRow & 2047;
#pragma unroll
                for (int a = 0; a < 4; ++a) {
                    int d = a * 16 + laneN;
                    vtO[((size_t)(bb * HEADS + h) * DHEAD + d) * SEQ + n] =
                        f2bf(acc[mt][a][r]);
                }
            }
    }
}

// ---------------- 4. Flash attention (WMMA bf16 + double-buffered TDM K/V) ----------------
__global__ __launch_bounds__(256)
void flash_attn(const unsigned short* __restrict__ qA,
                const unsigned short* __restrict__ kA,
                const unsigned short* __restrict__ vtA,
                unsigned short* __restrict__ outA)   // [b,n,1024] bf16
{
    __shared__ unsigned short kT[2][KTILE * DHEAD];  // [key][d]
    __shared__ unsigned short vT[2][DHEAD * KTILE];  // [d][key]
    __shared__ unsigned short pT[8][16 * 32];        // per-wave P tile
    int wave  = threadIdx.x >> 5;
    int lane  = threadIdx.x & 31;
    int laneN = lane & 15;
    int hi    = lane >> 4;
    int bh    = blockIdx.x;            // 0..63
    int b     = bh >> 4, h = bh & 15;
    int qbase = blockIdx.y * 128 + wave * 16;

    const unsigned short* qBase = qA  + (size_t)bh * SEQ * DHEAD;
    const unsigned short* kBase = kA  + (size_t)bh * SEQ * DHEAD;
    const unsigned short* vBase = vtA + (size_t)bh * DHEAD * SEQ;

    // Q A-fragments, pre-scaled by DHEAD^-0.5 = 0.125 (exact)
    FragBF fq[2];
    {
        const unsigned short* qr = qBase + (size_t)(qbase + laneN) * DHEAD;
#pragma unroll
        for (int s = 0; s < 2; ++s) {
            v8us t0 = *(const v8us*)(qr + s * 32 + hi * 8);
            v8us t1 = *(const v8us*)(qr + s * 32 + 16 + hi * 8);
#pragma unroll
            for (int i = 0; i < 8; ++i) {
                fq[s].u[i]     = f2bf(bf2f(t0[i]) * 0.125f);
                fq[s].u[i + 8] = f2bf(bf2f(t1[i]) * 0.125f);
            }
        }
    }

    float m[8], l[8];
    v8f o[4] = {};
#pragma unroll
    for (int r = 0; r < 8; ++r) { m[r] = -3.0e38f; l[r] = 0.f; }

    unsigned short* Pt = pT[wave];
    const int NT = SEQ / KTILE;        // 32 key tiles

#ifdef HAVE_TDM
    if (wave == 0) {                   // prologue: issue tile 0
        tdm_load_2d(kBase, lds_addr_of(kT[0]), DHEAD, KTILE, DHEAD);
        tdm_load_2d(vBase, lds_addr_of(vT[0]), KTILE, DHEAD, SEQ);
    }
#endif

    for (int it = 0; it < NT; ++it) {
        int cur = it & 1;
        int kb  = it * KTILE;
#ifdef HAVE_TDM
        if (wave == 0) {
            if (it + 1 < NT) {
                int nkb = (it + 1) * KTILE;
                tdm_load_2d(kBase + (size_t)nkb * DHEAD, lds_addr_of(kT[1 - cur]),
                            DHEAD, KTILE, DHEAD);
                tdm_load_2d(vBase + nkb, lds_addr_of(vT[1 - cur]),
                            KTILE, DHEAD, SEQ);
                wait_tensorcnt_2();    // TDM in-order: tile `it` complete
            } else {
                wait_tensorcnt_0();
            }
        }
        __syncthreads();               // tile `it` visible to all waves
#else
        __syncthreads();               // previous readers of kT[cur] done
        for (int i = threadIdx.x; i < (KTILE * DHEAD) / 8; i += 256)
            *(v8us*)(kT[cur] + i * 8) =
                *(const v8us*)(kBase + (size_t)kb * DHEAD + i * 8);
        for (int i = threadIdx.x; i < (DHEAD * KTILE) / 8; i += 256) {
            int d = i >> 3, c = (i & 7) * 8;
            *(v8us*)(vT[cur] + d * KTILE + c) =
                *(const v8us*)(vBase + (size_t)d * SEQ + kb + c);
        }
        __syncthreads();
#endif
        const unsigned short* kTile = kT[cur];
        const unsigned short* vTile = vT[cur];

#pragma unroll
        for (int t2 = 0; t2 < 2; ++t2) {
            int k0 = t2 * 32;
            // S = Q * K^T : two 16-key tiles, each chained over d (2 WMMAs)
            v8f S[2] = {};
#pragma unroll
            for (int t = 0; t < 2; ++t) {
                const unsigned short* kr =
                    kTile + (k0 + t * 16 + laneN) * DHEAD + hi * 8;
#pragma unroll
                for (int s = 0; s < 2; ++s) {
                    FragBF fk;
                    fk.h[0] = *(const v8us*)(kr + s * 32);
                    fk.h[1] = *(const v8us*)(kr + s * 32 + 16);
                    S[t] = __builtin_amdgcn_wmma_f32_16x16x32_bf16(
                        false, fq[s].bf, false, fk.bf, (short)0, S[t], false, false);
                }
            }

            // online softmax per row (row = hi*8 + r, spread over 16 lanes)
            float sc[8];
#pragma unroll
            for (int r = 0; r < 8; ++r) {
                float mx = fmaxf(S[0][r], S[1][r]);
#pragma unroll
                for (int off = 8; off >= 1; off >>= 1)
                    mx = fmaxf(mx, __shfl_xor(mx, off, 32));
                float mn = fmaxf(m[r], mx);
                float p0 = __expf(S[0][r] - mn);
                float p1 = __expf(S[1][r] - mn);
                float rs = p0 + p1;
#pragma unroll
                for (int off = 8; off >= 1; off >>= 1)
                    rs += __shfl_xor(rs, off, 32);
                float corr = __expf(m[r] - mn);
                l[r] = l[r] * corr + rs;
                m[r] = mn;
                sc[r] = corr;
                Pt[(hi * 8 + r) * 32 + laneN]      = f2bf(p0);
                Pt[(hi * 8 + r) * 32 + 16 + laneN] = f2bf(p1);
            }
#pragma unroll
            for (int a = 0; a < 4; ++a)
#pragma unroll
                for (int r = 0; r < 8; ++r)
                    o[a][r] *= sc[r];

            asm volatile("s_wait_dscnt 0" ::: "memory");

            FragBF fp;
            fp.h[0] = *(const v8us*)(Pt + laneN * 32 + hi * 8);
            fp.h[1] = *(const v8us*)(Pt + laneN * 32 + 16 + hi * 8);

            // O += P * V  (V^T rows in LDS are contiguous in key dim)
#pragma unroll
            for (int a = 0; a < 4; ++a) {
                const unsigned short* vr =
                    vTile + (a * 16 + laneN) * KTILE + k0 + hi * 8;
                FragBF fv;
                fv.h[0] = *(const v8us*)(vr);
                fv.h[1] = *(const v8us*)(vr + 16);
                o[a] = __builtin_amdgcn_wmma_f32_16x16x32_bf16(
                    false, fp.bf, false, fv.bf, (short)0, o[a], false, false);
            }
        }
#ifdef HAVE_TDM
        __syncthreads();   // all waves done with kT/vT[cur] before it's re-staged
#endif
    }

    // finalize: divide by l, write [b, n, h*64+d] bf16
#pragma unroll
    for (int r = 0; r < 8; ++r) {
        int n = qbase + hi * 8 + r;
        float inv = 1.f / l[r];
        unsigned short* orow = outA + ((size_t)(b * SEQ + n)) * INNER + h * DHEAD;
#pragma unroll
        for (int a = 0; a < 4; ++a)
            orow[a * 16 + laneN] = f2bf(o[a][r] * inv);
    }
}

// ---------------- 5. Output projection GEMM (64Mx64N/wave) + bias -> fp32 ----------------
__global__ __launch_bounds__(128)
void out_gemm(const unsigned short* __restrict__ A,   // attn_out bf16 [8192][1024]
              const unsigned short* __restrict__ Bt,  // woutT bf16 [1024][1024]
              const float* __restrict__ bias,
              float* __restrict__ out)
{
    int wave  = threadIdx.x >> 5;
    int lane  = threadIdx.x & 31;
    int laneN = lane & 15;
    int hi    = lane >> 4;
    int m0    = blockIdx.y * 64;
    int nb    = blockIdx.x * 256 + wave * 64;

    v8f acc[4][4] = {};
    const unsigned short* arow[4];
#pragma unroll
    for (int mt = 0; mt < 4; ++mt)
        arow[mt] = A + (size_t)(m0 + mt * 16 + laneN) * DIM;
    const unsigned short* brow[4];
#pragma unroll
    for (int a = 0; a < 4; ++a)
        brow[a] = Bt + (size_t)(nb + a * 16 + laneN) * DIM + hi * 8;
    int aofs = hi * 8;

    for (int k = 0; k < DIM; k += 32) {
        FragBF fa[4];
#pragma unroll
        for (int mt = 0; mt < 4; ++mt) {
            fa[mt].h[0] = *(const v8us*)(arow[mt] + k + aofs);
            fa[mt].h[1] = *(const v8us*)(arow[mt] + k + 16 + aofs);
        }
#pragma unroll
        for (int a = 0; a < 4; ++a) {
            FragBF fb;
            fb.h[0] = *(const v8us*)(brow[a] + k);
            fb.h[1] = *(const v8us*)(brow[a] + k + 16);
#pragma unroll
            for (int mt = 0; mt < 4; ++mt)
                acc[mt][a] = __builtin_amdgcn_wmma_f32_16x16x32_bf16(
                    false, fa[mt].bf, false, fb.bf, (short)0, acc[mt][a], false, false);
        }
    }
#pragma unroll
    for (int mt = 0; mt < 4; ++mt)
#pragma unroll
        for (int r = 0; r < 8; ++r) {
            int g = m0 + mt * 16 + hi * 8 + r;
            float* orow = out + (size_t)g * DIM;
#pragma unroll
            for (int a = 0; a < 4; ++a) {
                int c = nb + a * 16 + laneN;
                orow[c] = acc[mt][a][r] + bias[c];
            }
        }
}

// ---------------- launch ----------------
extern "C" void kernel_launch(void* const* d_in, const int* in_sizes, int n_in,
                              void* d_out, int out_size, void* d_ws, size_t ws_size,
                              hipStream_t stream) {
    (void)in_sizes; (void)n_in; (void)out_size; (void)ws_size;
    const float* x     = (const float*)d_in[0];
    const float* ln_g  = (const float*)d_in[1];
    const float* ln_b  = (const float*)d_in[2];
    const float* w_qkv = (const float*)d_in[3];
    const float* w_out = (const float*)d_in[4];
    const float* b_out = (const float*)d_in[5];
    float* out = (float*)d_out;

    size_t off = 0;
    auto alloc = [&](size_t bytes) -> void* {
        void* p = (char*)d_ws + off;
        off += (bytes + 255) & ~(size_t)255;
        return p;
    };
    unsigned short* xn    = (unsigned short*)alloc((size_t)ROWS * DIM * 2);
    unsigned short* wqkvT = (unsigned short*)alloc((size_t)NQKV * DIM * 2);
    unsigned short* woutT = (unsigned short*)alloc((size_t)DIM * DIM * 2);
    float* cosT           = (float*)alloc((size_t)SEQ * 32 * 4);
    float* sinT           = (float*)alloc((size_t)SEQ * 32 * 4);
    unsigned short* qB    = (unsigned short*)alloc((size_t)BATCH * HEADS * SEQ * DHEAD * 2);
    unsigned short* kB    = (unsigned short*)alloc((size_t)BATCH * HEADS * SEQ * DHEAD * 2);
    unsigned short* vtB   = (unsigned short*)alloc((size_t)BATCH * HEADS * SEQ * DHEAD * 2);
    unsigned short* aoB   = (unsigned short*)alloc((size_t)ROWS * INNER * 2);

    ln_kernel<<<ROWS, 256, 0, stream>>>(x, ln_g, ln_b, xn);
    conv_wqkv<<<(NQKV * DIM) / 256, 256, 0, stream>>>(w_qkv, wqkvT);
    conv_wout<<<(DIM * DIM) / 256, 256, 0, stream>>>(w_out, woutT);
    rope_tables_kernel<<<(SEQ * 32) / 256, 256, 0, stream>>>(cosT, sinT);

    qkv_gemm<<<dim3(NQKV / 256, ROWS / 64), 128, 0, stream>>>(
        xn, wqkvT, cosT, sinT, qB, kB, vtB);

    flash_attn<<<dim3(BATCH * HEADS, SEQ / 128), 256, 0, stream>>>(
        qB, kB, vtB, aoB);

    out_gemm<<<dim3(DIM / 256, ROWS / 64), 128, 0, stream>>>(
        aoB, woutT, b_out, out);
}
